// GATTCG_6803228197458
// MI455X (gfx1250) — matrix operand order repaired and checked by reading, not compiled, not run on hardware
//
#include <hip/hip_runtime.h>

// ---------------------------------------------------------------------------
// MI455X / gfx1250 implementation of the GAT+LSTM graph network forward pass.
// Dense GEMMs (LSTM recurrence, attention-FC, GAT lin_src) use
// v_wmma_f32_16x16x32_f16 (f16 inputs, f32 accumulate). WMMA loops are
// tile-outer / kc-inner with a single live accumulator (no spills).
// Transcendentals use the CDNA5 TRANS pipe (v_rcp_f32 / v_tanh_f32).
// ---------------------------------------------------------------------------

typedef unsigned int u32;
typedef _Float16 v16h __attribute__((ext_vector_type(16)));
typedef float    v8f  __attribute__((ext_vector_type(8)));

#define LAGT  10
#define IND   12
#define HID   256
#define GATES 1024
#define HC    512

__device__ inline v8f vzero() {
  v8f z;
#pragma unroll
  for (int k = 0; k < 8; ++k) z[k] = 0.f;
  return z;
}
__device__ inline float frcp(float x) { return __builtin_amdgcn_rcpf(x); }
__device__ inline float sigf(float x) { return frcp(1.f + __expf(-x)); }
__device__ inline float ftanh(float x) {
#if __has_builtin(__builtin_amdgcn_tanhf)
  return __builtin_amdgcn_tanhf(x);      // gfx1250 v_tanh_f32
#else
  float e = __expf(2.f * x);
  return 1.f - 2.f * frcp(e + 1.f);
#endif
}

// ordered encoding so unsigned atomicMax == float max
__device__ inline u32 fenc(float f) {
  u32 u = __float_as_uint(f);
  return (u >> 31) ? ~u : (u | 0x80000000u);
}
__device__ inline float fdec(u32 u) {
  return (u & 0x80000000u) ? __uint_as_float(u & 0x7fffffffu) : __uint_as_float(~u);
}

union AFrag { v16h v; u32 u[8]; };
union BFrag { v16h v; uint4 q[2]; };

// ---------------------------------------------------------------------------
// Pack weight W[Nout,Kin] into WMMA-B layout (B = W^T, K padded to KC*32).
// dword index = ((kc*NT + nt)*32 + lane)*8 + r ; K = (r>>2)*16 + (lane>>4)*8
// + 2*(r&3) (+1) ; n = nt*16 + (lane&15)
// ---------------------------------------------------------------------------
__global__ void k_pack_b(const float* __restrict__ W, u32* __restrict__ out,
                         int Nout, int Kin, int KC) {
  long idx = (long)blockIdx.x * 256 + threadIdx.x;
  long total = (long)KC * (Nout / 16) * 256;
  if (idx >= total) return;
  int r    = (int)(idx & 7);
  int lane = (int)((idx >> 3) & 31);
  long t2  = idx >> 8;
  int NT   = Nout / 16;
  int nt   = (int)(t2 % NT);
  int kc   = (int)(t2 / NT);
  int n    = nt * 16 + (lane & 15);
  int k0   = kc * 32 + ((r >> 2) * 16) + ((lane >> 4) * 8) + 2 * (r & 3);
  float w0 = (k0 < Kin)     ? W[(long)n * Kin + k0]     : 0.f;
  float w1 = (k0 + 1 < Kin) ? W[(long)n * Kin + k0 + 1] : 0.f;
  union { _Float16 h[2]; u32 u; } pk;
  pk.h[0] = (_Float16)w0; pk.h[1] = (_Float16)w1;
  out[idx] = pk.u;
}

__global__ void k_zero(float* p, long n) {
  long g = (long)blockIdx.x * 256 + threadIdx.x;
  if (g < n) p[g] = 0.f;
}

// ---------------------------------------------------------------------------
// Collapse (lin_edge,att_edge)->E9[9][4] and (lin_macro,att_macro)->M3[3][4]
// e9m layout: [0..35]=E9_1, [36..47]=M3_1, [48..83]=E9_2, [84..95]=M3_2
// ---------------------------------------------------------------------------
__global__ void k_prep(const float* le1, const float* ae1, const float* lm1, const float* am1,
                       const float* le2, const float* ae2, const float* lm2, const float* am2,
                       float* e9m) {
  int tid = threadIdx.x;
  if (tid < 36) {
    int j = tid / 4, h = tid % 4;
    float s = 0.f;
    for (int c = 0; c < 128; ++c) s += le1[(h * 128 + c) * 9 + j] * ae1[h * 128 + c];
    e9m[tid] = s;
  } else if (tid < 48) {
    int t2 = tid - 36; int j = t2 / 4, h = t2 % 4;
    float s = 0.f;
    for (int c = 0; c < 128; ++c) s += lm1[(h * 128 + c) * 3 + j] * am1[h * 128 + c];
    e9m[tid] = s;
  } else if (tid < 84) {
    int t2 = tid - 48; int j = t2 / 4, h = t2 % 4;
    float s = 0.f;
    for (int c = 0; c < 128; ++c) s += le2[(h * 128 + c) * 9 + j] * ae2[h * 128 + c];
    e9m[tid] = s;
  } else if (tid < 96) {
    int t2 = tid - 84; int j = t2 / 4, h = t2 % 4;
    float s = 0.f;
    for (int c = 0; c < 128; ++c) s += lm2[(h * 128 + c) * 3 + j] * am2[h * 128 + c];
    e9m[tid] = s;
  }
}

// ---------------------------------------------------------------------------
// LSTM + online-softmax time attention + FC(relu). 16 nodes/block, 8 waves.
// Recurrent GEMM h[16,256] @ Whh^T[256,1024] via WMMA each step.
// Gates bias+x part preloaded into the WMMA accumulator (C layout).
// ---------------------------------------------------------------------------
__global__ __launch_bounds__(256)
void k_lstm(const float* __restrict__ x0, const float* __restrict__ Wih,
            const float* __restrict__ bih, const float* __restrict__ bhh,
            const u32* __restrict__ Bwhh, const float* __restrict__ tattW,
            const u32* __restrict__ Blfc, const float* __restrict__ lfc_b,
            float* __restrict__ xenc) {
  __shared__ float    s_gl[16][GATES + 16];
  __shared__ float    s_h[16][HID + 4];
  __shared__ float    s_c[16][HID + 4];
  __shared__ _Float16 s_ha[16][HID + 8];
  __shared__ float    s_x[16][LAGT * IND];
  __shared__ float    s_pool[16][HID + 4];
  __shared__ float    s_bias[GATES];
  __shared__ float    s_red[16][17];
  __shared__ float    s_m[16], s_z[16], s_scl[16], s_wt[16];

  const int tid = threadIdx.x;
  const int lane = tid & 31, wv = tid >> 5;
  const int node0 = blockIdx.x * 16;
  const int nrow = tid >> 4;   // node for elementwise work
  const int jgrp = tid & 15;   // 16 threads per node

  for (int i = tid; i < 16 * LAGT * IND; i += 256) {
    int n = i / (LAGT * IND), r = i % (LAGT * IND);
    s_x[n][r] = x0[(long)(node0 + n) * (LAGT * IND) + r];
  }
  for (int i = tid; i < 16 * HID; i += 256) {
    int n = i >> 8, hh = i & 255;
    s_h[n][hh] = 0.f; s_c[n][hh] = 0.f;
    s_ha[n][hh] = (_Float16)0.f; s_pool[n][hh] = 0.f;
  }
  for (int i = tid; i < GATES; i += 256) s_bias[i] = bih[i] + bhh[i];
  if (tid < 16) { s_m[tid] = -1e30f; s_z[tid] = 0.f; }
  __syncthreads();

  const int rr = lane & 15, hi = lane >> 4;

  for (int t = 0; t < LAGT; ++t) {
    // gates = bias + x_t @ Wih^T   (thread covers 64 cols of one node)
    {
      float xv[IND];
#pragma unroll
      for (int d = 0; d < IND; ++d) xv[d] = s_x[nrow][t * IND + d];
      for (int q = 0; q < 64; ++q) {
        int col = jgrp * 64 + q;
        float g = s_bias[col];
#pragma unroll
        for (int d = 0; d < IND; ++d) g += xv[d] * Wih[col * IND + d];
        s_gl[nrow][col] = g;
      }
    }
    __syncthreads();

    // gates += h @ Whh^T : wave wv owns output cols [wv*128, wv*128+128)
#pragma unroll 1
    for (int i = 0; i < 8; ++i) {
      const int col = (wv * 8 + i) * 16 + rr;
      const int rbase = (lane < 16) ? 0 : 8;
      v8f acc;                              // preload bias+x part (C layout)
#pragma unroll
      for (int r = 0; r < 8; ++r) acc[r] = s_gl[rbase + r][col];
#pragma unroll 2
      for (int kc = 0; kc < 8; ++kc) {
        AFrag af;
#pragma unroll
        for (int r = 0; r < 4; ++r) {
          af.u[r]     = *(const u32*)&s_ha[rr][kc * 32 + hi * 8 + 2 * r];
          af.u[4 + r] = *(const u32*)&s_ha[rr][kc * 32 + 16 + hi * 8 + 2 * r];
        }
        const uint4* bp = (const uint4*)(Bwhh + (((long)kc * 64 + (wv * 8 + i)) * 32 + lane) * 8);
        __builtin_prefetch((const char*)bp + 64 * 1024, 0, 1);   // next kc line
        BFrag bf; bf.q[0] = bp[0]; bf.q[1] = bp[1];
        acc = __builtin_amdgcn_wmma_f32_16x16x32_f16(
            false, af.v, false, bf.v, (short)0, acc, false, false);
      }
#pragma unroll
      for (int r = 0; r < 8; ++r) s_gl[rbase + r][col] = acc[r];
    }
    __syncthreads();

    // LSTM cell (i,f,g,o gate order) + time-attention logit partial
    float part = 0.f;
    for (int q = 0; q < 16; ++q) {
      int hh = jgrp * 16 + q;
      float ig = s_gl[nrow][hh], fg = s_gl[nrow][HID + hh];
      float gg = s_gl[nrow][2 * HID + hh], og = s_gl[nrow][3 * HID + hh];
      float c = sigf(fg) * s_c[nrow][hh] + sigf(ig) * ftanh(gg);
      float h = sigf(og) * ftanh(c);
      s_c[nrow][hh] = c; s_h[nrow][hh] = h; s_ha[nrow][hh] = (_Float16)h;
      part += h * tattW[t * HID + hh];
    }
    s_red[nrow][jgrp] = part;
    __syncthreads();
    if (jgrp == 0) {
      float l = 0.f;
      for (int k = 0; k < 16; ++k) l += s_red[nrow][k];
      float m = s_m[nrow], mn = fmaxf(m, l);
      float scl = __expf(m - mn), wt = __expf(l - mn);
      s_z[nrow] = s_z[nrow] * scl + wt;
      s_m[nrow] = mn; s_scl[nrow] = scl; s_wt[nrow] = wt;
    }
    __syncthreads();
    for (int q = 0; q < 16; ++q) {
      int hh = jgrp * 16 + q;
      s_pool[nrow][hh] = s_pool[nrow][hh] * s_scl[nrow] + s_wt[nrow] * s_h[nrow][hh];
    }
    __syncthreads();
  }

  // pooled -> f16 A matrix for FC
  {
    float zi = frcp(s_z[nrow]);
    for (int q = 0; q < 16; ++q) {
      int hh = jgrp * 16 + q;
      s_ha[nrow][hh] = (_Float16)(s_pool[nrow][hh] * zi);
    }
  }
  __syncthreads();

  // FC 256->256, relu ; wave wv owns cols [wv*32, wv*32+32)
#pragma unroll 1
  for (int i = 0; i < 2; ++i) {
    v8f acc = vzero();
#pragma unroll 2
    for (int kc = 0; kc < 8; ++kc) {
      AFrag af;
#pragma unroll
      for (int r = 0; r < 4; ++r) {
        af.u[r]     = *(const u32*)&s_ha[rr][kc * 32 + hi * 8 + 2 * r];
        af.u[4 + r] = *(const u32*)&s_ha[rr][kc * 32 + 16 + hi * 8 + 2 * r];
      }
      const uint4* bp = (const uint4*)(Blfc + (((long)kc * 16 + (wv * 2 + i)) * 32 + lane) * 8);
      BFrag bf; bf.q[0] = bp[0]; bf.q[1] = bp[1];
      acc = __builtin_amdgcn_wmma_f32_16x16x32_f16(
          false, af.v, false, bf.v, (short)0, acc, false, false);
    }
    int col = (wv * 2 + i) * 16 + rr;
    int rbase = (lane < 16) ? 0 : 8;
#pragma unroll
    for (int r = 0; r < 8; ++r)
      xenc[(long)(node0 + rbase + r) * HID + col] = fmaxf(acc[r] + lfc_b[col], 0.f);
  }
}

// ---------------------------------------------------------------------------
// Macro RNN (8-dim tanh) + online time attention + FC(relu) -> a_mac per layer
// ---------------------------------------------------------------------------
__global__ __launch_bounds__(256)
void k_macro(const float* __restrict__ mseq, const float* __restrict__ Wih,
             const float* __restrict__ Whh, const float* __restrict__ bi,
             const float* __restrict__ bh, const float* __restrict__ attW,
             const float* __restrict__ fcW, const float* __restrict__ fcb,
             const float* __restrict__ e9m, float* __restrict__ aM1,
             float* __restrict__ aM2, int E_) {
  __shared__ float w_ih[8], w_hh[64], w_b[8], w_att[80], w_fc[24], w_fcb[3], w_m1[12], w_m2[12];
  int tid = threadIdx.x;
  if (tid < 8)  { w_ih[tid] = Wih[tid]; w_b[tid] = bi[tid] + bh[tid]; }
  if (tid < 64) w_hh[tid] = Whh[tid];
  if (tid < 80) w_att[tid] = attW[tid];
  if (tid < 24) w_fc[tid] = fcW[tid];
  if (tid < 3)  w_fcb[tid] = fcb[tid];
  if (tid < 12) { w_m1[tid] = e9m[36 + tid]; w_m2[tid] = e9m[84 + tid]; }
  __syncthreads();
  long e = (long)blockIdx.x * 256 + tid;
  if (e >= E_) return;
  float h[8], pool[8], m = -1e30f, z = 0.f;
#pragma unroll
  for (int j = 0; j < 8; ++j) { h[j] = 0.f; pool[j] = 0.f; }
  for (int t = 0; t < LAGT; ++t) {
    float xt = mseq[e * LAGT + t];
    float hn[8];
#pragma unroll
    for (int j = 0; j < 8; ++j) {
      float g = w_ih[j] * xt + w_b[j];
#pragma unroll
      for (int k = 0; k < 8; ++k) g += w_hh[j * 8 + k] * h[k];
      hn[j] = ftanh(g);
    }
    float l = 0.f;
#pragma unroll
    for (int j = 0; j < 8; ++j) l += hn[j] * w_att[t * 8 + j];
    float mn = fmaxf(m, l), scl = __expf(m - mn), wt = __expf(l - mn);
    z = z * scl + wt;
#pragma unroll
    for (int j = 0; j < 8; ++j) { pool[j] = pool[j] * scl + wt * hn[j]; h[j] = hn[j]; }
    m = mn;
  }
  float mo[3];
  float zi = frcp(z);
#pragma unroll
  for (int c = 0; c < 3; ++c) {
    float s = w_fcb[c];
#pragma unroll
    for (int j = 0; j < 8; ++j) s += (pool[j] * zi) * w_fc[c * 8 + j];
    mo[c] = fmaxf(s, 0.f);
  }
#pragma unroll
  for (int hh = 0; hh < 4; ++hh) {
    float a = 0.f, b = 0.f;
#pragma unroll
    for (int c = 0; c < 3; ++c) { a += mo[c] * w_m1[c * 4 + hh]; b += mo[c] * w_m2[c * 4 + hh]; }
    aM1[e * 4 + hh] = a; aM2[e * 4 + hh] = b;
  }
}

// ---------------------------------------------------------------------------
// Edge conv: 3x9x9 -> conv(16,3,3,s2) -> relu -> maxpool2 -> 64 -> 9
// One wave per edge; also accumulates per-dst mean stats and a_edge dots.
// ---------------------------------------------------------------------------
__global__ __launch_bounds__(256)
void k_edgeconv(const float* __restrict__ eattr, const int* __restrict__ dst,
                const float* __restrict__ convW, const float* __restrict__ convB,
                const float* __restrict__ outW, const float* __restrict__ outB,
                const float* __restrict__ e9m, float* __restrict__ aE1,
                float* __restrict__ aE2, float* __restrict__ loop_sum,
                float* __restrict__ cnt, int E_) {
  __shared__ float s_x[8][244];
  __shared__ float s_w[432];
  __shared__ float s_b[16];
  __shared__ float s_ow[576];
  __shared__ float s_ob[9];
  __shared__ float s_pool[8][64];
  __shared__ float s_eo[8][9];
  int tid = threadIdx.x, lane = tid & 31, we = tid >> 5;
  long e = (long)blockIdx.x * 8 + we;
  for (int i = tid; i < 432; i += 256) s_w[i] = convW[i];
  for (int i = tid; i < 576; i += 256) s_ow[i] = outW[i];
  if (tid < 16) s_b[tid] = convB[tid];
  if (tid < 9)  s_ob[tid] = outB[tid];
  for (int i = lane; i < 243; i += 32) s_x[we][i] = eattr[e * 243 + i];
  __syncthreads();

  int oc = lane >> 1, half = lane & 1;
  float pl0 = -1e30f, pl1 = -1e30f;
  for (int oy = 2 * half; oy < 2 * half + 2; ++oy)
    for (int ox = 0; ox < 4; ++ox) {
      float a = s_b[oc];
#pragma unroll
      for (int ic = 0; ic < 3; ++ic)
#pragma unroll
        for (int ky = 0; ky < 3; ++ky)
#pragma unroll
          for (int kx = 0; kx < 3; ++kx)
            a += s_x[we][ic * 81 + (2 * oy + ky) * 9 + (2 * ox + kx)] *
                 s_w[((oc * 3 + ic) * 3 + ky) * 3 + kx];
      a = fmaxf(a, 0.f);
      if ((ox >> 1) == 0) pl0 = fmaxf(pl0, a); else pl1 = fmaxf(pl1, a);
    }
  s_pool[we][oc * 4 + half * 2 + 0] = pl0;
  s_pool[we][oc * 4 + half * 2 + 1] = pl1;
  __syncthreads();

  if (lane < 9) {
    float o = s_ob[lane];
#pragma unroll
    for (int k = 0; k < 64; ++k) o += s_pool[we][k] * s_ow[lane * 64 + k];
    s_eo[we][lane] = o;
    atomicAdd(&loop_sum[(long)dst[e] * 9 + lane], o);
  } else if (lane == 9) {
    atomicAdd(&cnt[dst[e]], 1.0f);
  }
  __syncthreads();
  if (lane < 4) {
    float a1 = 0.f, a2 = 0.f;
#pragma unroll
    for (int j = 0; j < 9; ++j) {
      float v = s_eo[we][j];
      a1 += v * e9m[j * 4 + lane];
      a2 += v * e9m[48 + j * 4 + lane];
    }
    aE1[e * 4 + lane] = a1; aE2[e * 4 + lane] = a2;
  }
}

// self-loop edge attr = mean of incoming; dot with E9 per layer
__global__ void k_loopattr(const float* __restrict__ loop_sum, const float* __restrict__ cnt,
                           const float* __restrict__ e9m, float* __restrict__ aL1,
                           float* __restrict__ aL2, int N_) {
  int n = blockIdx.x * 256 + threadIdx.x;
  if (n >= N_) return;
  float ci = frcp(fmaxf(cnt[n], 1.f));
  float la[9];
#pragma unroll
  for (int j = 0; j < 9; ++j) la[j] = loop_sum[(long)n * 9 + j] * ci;
#pragma unroll
  for (int h = 0; h < 4; ++h) {
    float a1 = 0.f, a2 = 0.f;
#pragma unroll
    for (int j = 0; j < 9; ++j) { a1 += la[j] * e9m[j * 4 + h]; a2 += la[j] * e9m[48 + j * 4 + h]; }
    aL1[n * 4 + h] = a1; aL2[n * 4 + h] = a2;
  }
}

// training-mode batchnorm stats (biased var) over rows, one column per block
__global__ __launch_bounds__(256)
void k_bnstats(const float* __restrict__ X, int rows, int cols,
               float* __restrict__ mu, float* __restrict__ rstd) {
  int c = blockIdx.x, tid = threadIdx.x;
  float s = 0.f, s2 = 0.f;
  for (int r = tid; r < rows; r += 256) {
    float v = X[(long)r * cols + c];
    s += v; s2 += v * v;
  }
  for (int o = 16; o; o >>= 1) { s += __shfl_down(s, o, 32); s2 += __shfl_down(s2, o, 32); }
  __shared__ float rs[8], rs2[8];
  if ((tid & 31) == 0) { rs[tid >> 5] = s; rs2[tid >> 5] = s2; }
  __syncthreads();
  if (tid == 0) {
    float t = 0.f, t2 = 0.f;
    for (int k = 0; k < 8; ++k) { t += rs[k]; t2 += rs2[k]; }
    float m = t / rows;
    mu[c] = m;
    rstd[c] = rsqrtf(t2 / rows - m * m + 1e-5f);
  }
}

// ---------------------------------------------------------------------------
// WMMA GEMM: xs = BN(A) @ lin_src^T  ([16 rows/block] x [512 cols], 4 waves)
// P==0: A = x_enc (K=256). P==1: A = concat(BN2(out1)[512], x0_last[12]) K=544pad
// Full A tile staged in LDS once; tile-outer / kc-inner; one live accumulator.
// ---------------------------------------------------------------------------
template <int P, int KC>
__global__ __launch_bounds__(128)
void k_gemm_xs(const float* __restrict__ A0, const float* __restrict__ xlast,
               const float* __restrict__ mu, const float* __restrict__ rstd,
               const float* __restrict__ gamma, const float* __restrict__ beta,
               const u32* __restrict__ Bpack, float* __restrict__ C) {
  __shared__ _Float16 s_a[16][KC * 32 + 8];
  const int tid = threadIdx.x, lane = tid & 31, wv = tid >> 5;
  const int row0 = blockIdx.x * 16;
  const int rr = lane & 15, hi = lane >> 4;

  for (int i = tid; i < 16 * KC * 32; i += 128) {
    int rw = i / (KC * 32), gk = i % (KC * 32);
    float v;
    if (P == 0) {
      v = (A0[(long)(row0 + rw) * 256 + gk] - mu[gk]) * rstd[gk] * gamma[gk] + beta[gk];
    } else {
      if (gk < 512)
        v = (A0[(long)(row0 + rw) * 512 + gk] - mu[gk]) * rstd[gk] * gamma[gk] + beta[gk];
      else if (gk < 524)
        v = xlast[(long)(row0 + rw) * (LAGT * IND) + (LAGT - 1) * IND + (gk - 512)];
      else
        v = 0.f;
    }
    s_a[rw][gk] = (_Float16)v;
  }
  __syncthreads();

#pragma unroll 1
  for (int i = 0; i < 8; ++i) {
    v8f acc = vzero();
#pragma unroll 2
    for (int kc = 0; kc < KC; ++kc) {
      AFrag af;
#pragma unroll
      for (int r = 0; r < 4; ++r) {
        af.u[r]     = *(const u32*)&s_a[rr][kc * 32 + hi * 8 + 2 * r];
        af.u[4 + r] = *(const u32*)&s_a[rr][kc * 32 + 16 + hi * 8 + 2 * r];
      }
      const uint4* bp = (const uint4*)(Bpack + (((long)kc * 32 + (wv * 8 + i)) * 32 + lane) * 8);
      __builtin_prefetch((const char*)bp + 32 * 1024, 0, 1);   // next kc line
      BFrag bf; bf.q[0] = bp[0]; bf.q[1] = bp[1];
      acc = __builtin_amdgcn_wmma_f32_16x16x32_f16(
          false, af.v, false, bf.v, (short)0, acc, false, false);
    }
    int col = (wv * 8 + i) * 16 + rr;
    int rbase = (lane < 16) ? 0 : 8;
#pragma unroll
    for (int r = 0; r < 8; ++r) C[(long)(row0 + rbase + r) * HC + col] = acc[r];
  }
}

// a_src/a_dst = sum_c xs[n,h,c]*att[h,c] ; one node per block, 512 threads
__global__ __launch_bounds__(512)
void k_attsd(const float* __restrict__ xs, const float* __restrict__ attS,
             const float* __restrict__ attD, float* __restrict__ aS,
             float* __restrict__ aD) {
  long n = blockIdx.x;
  int j = threadIdx.x;
  float x = xs[n * HC + j];
  float ps = x * attS[j], pd = x * attD[j];
  for (int o = 16; o; o >>= 1) { ps += __shfl_down(ps, o, 32); pd += __shfl_down(pd, o, 32); }
  __shared__ float ss[16], sd[16];
  int wv = j >> 5;
  if ((j & 31) == 0) { ss[wv] = ps; sd[wv] = pd; }
  __syncthreads();
  if (j < 4) {
    float a = 0.f, b = 0.f;
    for (int k = 0; k < 4; ++k) { a += ss[j * 4 + k]; b += sd[j * 4 + k]; }
    aS[n * 4 + j] = a; aD[n * 4 + j] = b;
  }
}

__device__ inline void edge_logit(long i, int E_, const int* src, const int* dst,
                                  const float* aS, const float* aD, const float* aE,
                                  const float* aL, const float* aM,
                                  int& s, int& d, float l[4]) {
  const float *pe, *pm;
  if (i < E_) { s = src[i]; d = dst[i]; pe = &aE[i * 4]; pm = &aM[i * 4]; }
  else        { s = d = (int)(i - E_); pe = &aL[(long)s * 4]; pm = &aM[0]; }
#pragma unroll
  for (int h = 0; h < 4; ++h) {
    float x = aS[(long)s * 4 + h] + aD[(long)d * 4 + h] + pe[h] + pm[h];
    l[h] = (x >= 0.f) ? x : 0.2f * x;   // leaky_relu
  }
}

__global__ void k_pass1(const int* src, const int* dst, const float* aS, const float* aD,
                        const float* aE, const float* aL, const float* aM,
                        u32* maxb, int E_, int N_) {
  long i = (long)blockIdx.x * 256 + threadIdx.x;
  if (i >= (long)E_ + N_) return;
  int s, d; float l[4];
  edge_logit(i, E_, src, dst, aS, aD, aE, aL, aM, s, d, l);
#pragma unroll
  for (int h = 0; h < 4; ++h) atomicMax(&maxb[(long)d * 4 + h], fenc(l[h]));
}

__global__ void k_pass2a(const int* src, const int* dst, const float* aS, const float* aD,
                         const float* aE, const float* aL, const float* aM,
                         const u32* maxb, float* denom, float* exb, int E_, int N_) {
  long i = (long)blockIdx.x * 256 + threadIdx.x;
  if (i >= (long)E_ + N_) return;
  int s, d; float l[4];
  edge_logit(i, E_, src, dst, aS, aD, aE, aL, aM, s, d, l);
#pragma unroll
  for (int h = 0; h < 4; ++h) {
    float ex = __expf(l[h] - fdec(maxb[(long)d * 4 + h]));
    atomicAdd(&denom[(long)d * 4 + h], ex);
    exb[i * 4 + h] = ex;
  }
}

// numer[d] += ex * xs[s] ; one thread per 4 features of one edge
__global__ void k_pass2b(const int* src, const int* dst, const float* __restrict__ exb,
                         const float* __restrict__ xs, float* __restrict__ numer,
                         int E_, int N_) {
  long g = (long)blockIdx.x * 256 + threadIdx.x;
  long tot = ((long)E_ + N_) * 128;
  if (g >= tot) return;
  long i = g >> 7;
  int q = (int)(g & 127);
  int s, d;
  if (i < E_) { s = src[i]; d = dst[i]; } else { s = d = (int)(i - E_); }
  float e = exb[i * 4 + (q >> 5)];
  float4 xv = ((const float4*)xs)[(long)s * 128 + q];
  long base = (long)d * HC + q * 4;
  atomicAdd(&numer[base + 0], e * xv.x);
  atomicAdd(&numer[base + 1], e * xv.y);
  atomicAdd(&numer[base + 2], e * xv.z);
  atomicAdd(&numer[base + 3], e * xv.w);
}

// out = numer/denom + bias   (in place on numer buffer)
__global__ void k_pass3(float* __restrict__ numer, const float* __restrict__ denom,
                        const float* __restrict__ bias, int N_) {
  long g = (long)blockIdx.x * 256 + threadIdx.x;
  if (g >= (long)N_ * HC) return;
  long n = g >> 9;
  int col = (int)(g & 511);
  numer[g] = numer[g] * frcp(denom[n * 4 + (col >> 7)]) + bias[col];
}

// relu(x @ lin_W^T + b) then log_softmax over 3 classes
__global__ void k_head(const float* __restrict__ x, const float* __restrict__ W,
                       const float* __restrict__ b, float* __restrict__ out, int N_) {
  long n = (long)blockIdx.x * 256 + threadIdx.x;
  if (n >= N_) return;
  float r[3];
#pragma unroll
  for (int c = 0; c < 3; ++c) {
    float acc = b[c];
    for (int j = 0; j < HC; ++j) acc += x[n * HC + j] * W[c * HC + j];
    r[c] = fmaxf(acc, 0.f);
  }
  float m = fmaxf(r[0], fmaxf(r[1], r[2]));
  float lse = m + logf(__expf(r[0] - m) + __expf(r[1] - m) + __expf(r[2] - m));
#pragma unroll
  for (int c = 0; c < 3; ++c) out[n * 3 + c] = r[c] - lse;
}

// ---------------------------------------------------------------------------
// Host launcher
// ---------------------------------------------------------------------------
extern "C" void kernel_launch(void* const* d_in, const int* in_sizes, int n_in,
                              void* d_out, int out_size, void* d_ws, size_t ws_size,
                              hipStream_t stream) {
  // inputs in setup_inputs() dict (insertion) order
  const float* x0    = (const float*)d_in[0];   // [N,10,12]
  const float* eattr = (const float*)d_in[1];   // [E,243]
  const float* mseq  = (const float*)d_in[2];   // [E,10,1]
  const int*   eidx  = (const int*)d_in[3];     // [2,E]
  const float* lstm_Wih = (const float*)d_in[4];
  const float* lstm_Whh = (const float*)d_in[5];
  const float* lstm_bih = (const float*)d_in[6];
  const float* lstm_bhh = (const float*)d_in[7];
  const float* tatt_W   = (const float*)d_in[8];
  const float* lfc_W    = (const float*)d_in[9];
  const float* lfc_b    = (const float*)d_in[10];
  const float* rnn_Wih  = (const float*)d_in[11];
  const float* rnn_Whh  = (const float*)d_in[12];
  const float* rnn_bih  = (const float*)d_in[13];
  const float* rnn_bhh  = (const float*)d_in[14];
  const float* mtatt_W  = (const float*)d_in[15];
  const float* mfc_W    = (const float*)d_in[16];
  const float* mfc_b    = (const float*)d_in[17];
  const float* g1_gamma = (const float*)d_in[18];
  const float* g1_beta  = (const float*)d_in[19];
  const float* g2_gamma = (const float*)d_in[20];
  const float* g2_beta  = (const float*)d_in[21];
  const float* conv_W   = (const float*)d_in[22];
  const float* conv_b   = (const float*)d_in[23];
  const float* out_W    = (const float*)d_in[24];
  const float* out_b    = (const float*)d_in[25];
  const float* lin_W    = (const float*)d_in[26];
  const float* lin_b    = (const float*)d_in[27];
  const float* a1_lin_src   = (const float*)d_in[28];
  const float* a1_lin_edge  = (const float*)d_in[29];
  const float* a1_lin_macro = (const float*)d_in[30];
  const float* a1_att_src   = (const float*)d_in[31];
  const float* a1_att_dst   = (const float*)d_in[32];
  const float* a1_att_edge  = (const float*)d_in[33];
  const float* a1_att_macro = (const float*)d_in[34];
  const float* a1_bias      = (const float*)d_in[35];
  const float* a2_lin_src   = (const float*)d_in[36];
  const float* a2_lin_edge  = (const float*)d_in[37];
  const float* a2_lin_macro = (const float*)d_in[38];
  const float* a2_att_src   = (const float*)d_in[39];
  const float* a2_att_dst   = (const float*)d_in[40];
  const float* a2_att_edge  = (const float*)d_in[41];
  const float* a2_att_macro = (const float*)d_in[42];
  const float* a2_bias      = (const float*)d_in[43];

  const int Nn = in_sizes[0] / (LAGT * IND);   // 20000
  const int Ee = in_sizes[1] / 243;            // 160000
  const int* src = eidx;
  const int* dst = eidx + Ee;

  // workspace arena (256B aligned slots)
  char* base = (char*)d_ws;
  size_t off = 0;
  auto alloc = [&](size_t bytes) -> void* {
    void* p = base + off;
    off = (off + bytes + 255) & ~(size_t)255;
    return p;
  };
  u32*   packWhh = (u32*)alloc(8  * 64 * 256 * 4);
  u32*   packLfc = (u32*)alloc(8  * 16 * 256 * 4);
  u32*   packA1  = (u32*)alloc(8  * 32 * 256 * 4);
  u32*   packA2  = (u32*)alloc(17 * 32 * 256 * 4);
  float* e9m     = (float*)alloc(96 * 4);
  float* xenc    = (float*)alloc((size_t)Nn * 256 * 4);
  float* aM1     = (float*)alloc((size_t)Ee * 4 * 4);
  float* aM2     = (float*)alloc((size_t)Ee * 4 * 4);
  float* aE1     = (float*)alloc((size_t)Ee * 4 * 4);
  float* aE2     = (float*)alloc((size_t)Ee * 4 * 4);
  float* loop_sum= (float*)alloc((size_t)Nn * 9 * 4);
  float* cnt     = (float*)alloc((size_t)Nn * 4);
  float* aL1     = (float*)alloc((size_t)Nn * 4 * 4);
  float* aL2     = (float*)alloc((size_t)Nn * 4 * 4);
  float* mu1     = (float*)alloc(256 * 4);
  float* rstd1   = (float*)alloc(256 * 4);
  float* mu2     = (float*)alloc(512 * 4);
  float* rstd2   = (float*)alloc(512 * 4);
  float* xs      = (float*)alloc((size_t)Nn * HC * 4);
  float* asrc    = (float*)alloc((size_t)Nn * 4 * 4);
  float* adst    = (float*)alloc((size_t)Nn * 4 * 4);
  u32*   maxb    = (u32*)alloc((size_t)Nn * 4 * 4);
  float* denom   = (float*)alloc((size_t)Nn * 4 * 4);
  float* exb     = (float*)alloc(((size_t)Ee + Nn) * 4 * 4);
  float* out1    = (float*)alloc((size_t)Nn * HC * 4);
  float* out2    = out1;   // reused after GEMM2 has consumed out1
  (void)ws_size; (void)n_in; (void)out_size;

  const long EN = (long)Ee + Nn;

  // weight packing for WMMA B operands
  k_pack_b<<<512, 256, 0, stream>>>(lstm_Whh, packWhh, 1024, 256, 8);
  k_pack_b<<<128, 256, 0, stream>>>(lfc_W,    packLfc, 256, 256, 8);
  k_pack_b<<<256, 256, 0, stream>>>(a1_lin_src, packA1, 512, 256, 8);
  k_pack_b<<<544, 256, 0, stream>>>(a2_lin_src, packA2, 512, 524, 17);
  k_prep<<<1, 96, 0, stream>>>(a1_lin_edge, a1_att_edge, a1_lin_macro, a1_att_macro,
                               a2_lin_edge, a2_att_edge, a2_lin_macro, a2_att_macro, e9m);

  k_zero<<<(int)(((long)Nn * 9 + 255) / 256), 256, 0, stream>>>(loop_sum, (long)Nn * 9);
  k_zero<<<(Nn + 255) / 256, 256, 0, stream>>>(cnt, Nn);

  // encoders
  k_lstm<<<Nn / 16, 256, 0, stream>>>(x0, lstm_Wih, lstm_bih, lstm_bhh, packWhh,
                                      tatt_W, packLfc, lfc_b, xenc);
  k_macro<<<(Ee + 255) / 256, 256, 0, stream>>>(mseq, rnn_Wih, rnn_Whh, rnn_bih, rnn_bhh,
                                                mtatt_W, mfc_W, mfc_b, e9m, aM1, aM2, Ee);
  k_edgeconv<<<Ee / 8, 256, 0, stream>>>(eattr, dst, conv_W, conv_b, out_W, out_b, e9m,
                                         aE1, aE2, loop_sum, cnt, Ee);
  k_loopattr<<<(Nn + 255) / 256, 256, 0, stream>>>(loop_sum, cnt, e9m, aL1, aL2, Nn);

  // ---- GAT layer 1 ----
  k_bnstats<<<256, 256, 0, stream>>>(xenc, Nn, 256, mu1, rstd1);
  k_gemm_xs<0, 8><<<Nn / 16, 128, 0, stream>>>(xenc, nullptr, mu1, rstd1,
                                               g1_gamma, g1_beta, packA1, xs);
  k_attsd<<<Nn, 512, 0, stream>>>(xs, a1_att_src, a1_att_dst, asrc, adst);
  k_zero<<<(int)(((long)Nn * 4 + 255) / 256), 256, 0, stream>>>((float*)maxb, (long)Nn * 4);
  k_zero<<<(int)(((long)Nn * 4 + 255) / 256), 256, 0, stream>>>(denom, (long)Nn * 4);
  k_zero<<<(int)(((long)Nn * HC + 255) / 256), 256, 0, stream>>>(out1, (long)Nn * HC);
  k_pass1<<<(int)((EN + 255) / 256), 256, 0, stream>>>(src, dst, asrc, adst, aE1, aL1, aM1,
                                                       maxb, Ee, Nn);
  k_pass2a<<<(int)((EN + 255) / 256), 256, 0, stream>>>(src, dst, asrc, adst, aE1, aL1, aM1,
                                                        maxb, denom, exb, Ee, Nn);
  k_pass2b<<<(int)((EN * 128 + 255) / 256), 256, 0, stream>>>(src, dst, exb, xs, out1, Ee, Nn);
  k_pass3<<<(int)(((long)Nn * HC + 255) / 256), 256, 0, stream>>>(out1, denom, a1_bias, Nn);

  // ---- GAT layer 2 ----
  k_bnstats<<<512, 256, 0, stream>>>(out1, Nn, 512, mu2, rstd2);
  k_gemm_xs<1, 17><<<Nn / 16, 128, 0, stream>>>(out1, x0, mu2, rstd2,
                                                g2_gamma, g2_beta, packA2, xs);
  k_attsd<<<Nn, 512, 0, stream>>>(xs, a2_att_src, a2_att_dst, asrc, adst);
  k_zero<<<(int)(((long)Nn * 4 + 255) / 256), 256, 0, stream>>>((float*)maxb, (long)Nn * 4);
  k_zero<<<(int)(((long)Nn * 4 + 255) / 256), 256, 0, stream>>>(denom, (long)Nn * 4);
  k_zero<<<(int)(((long)Nn * HC + 255) / 256), 256, 0, stream>>>(out2, (long)Nn * HC);
  k_pass1<<<(int)((EN + 255) / 256), 256, 0, stream>>>(src, dst, asrc, adst, aE2, aL2, aM2,
                                                       maxb, Ee, Nn);
  k_pass2a<<<(int)((EN + 255) / 256), 256, 0, stream>>>(src, dst, asrc, adst, aE2, aL2, aM2,
                                                        maxb, denom, exb, Ee, Nn);
  k_pass2b<<<(int)((EN * 128 + 255) / 256), 256, 0, stream>>>(src, dst, exb, xs, out2, Ee, Nn);
  k_pass3<<<(int)(((long)Nn * HC + 255) / 256), 256, 0, stream>>>(out2, denom, a2_bias, Nn);

  // head: relu(linear) + log_softmax
  k_head<<<(Nn + 255) / 256, 256, 0, stream>>>(out2, lin_W, lin_b, (float*)d_out, Nn);
}
